// Decouple_76879914598841
// MI455X (gfx1250) — compile-verified, matrix operand order checked by reading, not run on hardware
//
#include <hip/hip_runtime.h>
#include <hip/hip_bf16.h>
#include <stdint.h>

// Problem constants
#define N_    4
#define C_    64
#define H_    192
#define W_    192
#define DIL   2
#define HP    194           // padded (pad=1 for the fused 3x3 conv)
#define WP    194
#define HW    (H_ * W_)     // 36864
#define K_TOT 576           // 64 ci * 9 taps
#define PLANE_ELEMS ((size_t)N_ * HP * WP * C_)   // 9,634,816 bf16 elems per plane

typedef __attribute__((ext_vector_type(16))) __bf16       v16bf;
typedef __attribute__((ext_vector_type(8)))  float        v8f;
typedef __attribute__((ext_vector_type(4)))  unsigned int u32x4;

union FragBF { v16bf bf; u32x4 q[2]; };

__device__ __forceinline__ unsigned short f2bf_rtne(float f) {
    unsigned int u = __float_as_uint(f);
    u += 0x7FFFu + ((u >> 16) & 1u);       // round to nearest even
    return (unsigned short)(u >> 16);
}
__device__ __forceinline__ float bf2f(unsigned short h) {
    return __uint_as_float(((unsigned int)h) << 16);
}

// ---------------------------------------------------------------------------
// Kernel 0: zero the padded intermediate planes (borders must be 0 for the
// implicit conv padding; harness poisons ws, so re-zero every call).
// ---------------------------------------------------------------------------
__global__ void zero_ws_kernel(u32x4* __restrict__ p, long long n16) {
    long long i = (long long)blockIdx.x * blockDim.x + threadIdx.x;
    if (i < n16) { u32x4 z = {0u, 0u, 0u, 0u}; p[i] = z; }
}

// ---------------------------------------------------------------------------
// Kernel 1: repack fuse_w [co][ci][ki][kj] f32 -> Wa[co][tap*64+ci] bf16
// split into hi + residual-lo planes (3xBF16 precision trick).
// ---------------------------------------------------------------------------
__global__ void repack_w_kernel(const float* __restrict__ fw,
                                unsigned short* __restrict__ wahi,
                                unsigned short* __restrict__ walo) {
    int t = blockIdx.x * blockDim.x + threadIdx.x;
    if (t >= C_ * K_TOT) return;
    int co  = t / K_TOT;
    int r   = t % K_TOT;
    int tap = r / C_;
    int ci  = r % C_;
    float w = fw[(co * C_ + ci) * 9 + tap];
    unsigned short hi = f2bf_rtne(w);
    wahi[t] = hi;
    walo[t] = f2bf_rtne(w - bf2f(hi));
}

// ---------------------------------------------------------------------------
// Kernel 2: dynamic 3x3 (dilation 2) per-pixel conv, fp32 accumulate.
// Block = 256 thr handles (n, h) x 32 w x all 64 channels.
// Phase 1: w-fastest lanes -> coalesced reads of y (the 340 MB stream) and x.
// Phase 2: LDS transpose -> c-fastest bf16 NHWC writes into padded buffer.
// ---------------------------------------------------------------------------
__global__ __launch_bounds__(256) void dynconv_kernel(
        const float* __restrict__ x, const float* __restrict__ y,
        unsigned short* __restrict__ s1hi, unsigned short* __restrict__ s1lo) {
    __shared__ float tile[C_][33];

    int b  = blockIdx.x;
    int wt = b % (W_ / 32);
    int h  = (b / (W_ / 32)) % H_;
    int n  = b / ((W_ / 32) * H_);
    int w0 = wt * 32;

    int tid = threadIdx.x;
    int wl  = tid & 31;          // w within tile
    int c0  = (tid >> 5) * 8;    // 8 channels per thread
    int w   = w0 + wl;

    const float* xb = x + (size_t)n * C_ * HW;
    const float* yb = y + (size_t)n * (C_ * 9) * HW;

    float acc[8];
    #pragma unroll
    for (int i = 0; i < 8; ++i) acc[i] = 0.0f;

    #pragma unroll
    for (int tap = 0; tap < 9; ++tap) {
        int row = h + (tap / 3) * DIL - DIL;
        int col = w + (tap % 3) * DIL - DIL;
        if (row < 0 || row >= H_ || col < 0 || col >= W_) continue;
        #pragma unroll
        for (int cc = 0; cc < 8; ++cc) {
            int c = c0 + cc;
            float xv = xb[(size_t)c * HW + row * W_ + col];
            float kv = yb[(size_t)(c * 9 + tap) * HW + h * W_ + w];
            acc[cc] += xv * kv;
        }
    }
    #pragma unroll
    for (int cc = 0; cc < 8; ++cc) tile[c0 + cc][wl] = acc[cc];
    __syncthreads();

    int c  = tid & 63;
    int wg = tid >> 6;           // 0..3
    #pragma unroll
    for (int it = 0; it < 8; ++it) {
        int wl2 = wg + it * 4;
        float v = tile[c][wl2];
        unsigned short hi = f2bf_rtne(v);
        unsigned short lo = f2bf_rtne(v - bf2f(hi));
        size_t a = (((size_t)n * HP + (h + 1)) * WP + (w0 + wl2 + 1)) * C_ + c;
        s1hi[a] = hi;
        s1lo[a] = lo;
    }
}

// ---------------------------------------------------------------------------
// Kernel 3: fused 3x3 conv (GEMM M=64, K=576) + bias + LeakyReLU via
// v_wmma_f32_16x16x32_bf16, 3-term bf16 split for ~fp32 accuracy.
// Block = 256 thr = 8 waves; each wave: 16 pixels x 4 co-tiles.
// Fragment loads: two b128 per operand per lane (channels contiguous in NHWC).
// ---------------------------------------------------------------------------
__global__ __launch_bounds__(256) void fuseconv_wmma_kernel(
        const unsigned short* __restrict__ s1hi, const unsigned short* __restrict__ s1lo,
        const unsigned short* __restrict__ wahi, const unsigned short* __restrict__ walo,
        const float* __restrict__ fb, float* __restrict__ out) {
    int b    = blockIdx.x;
    int p0   = b * 128;
    int n    = p0 / HW;
    int prem = p0 % HW;          // 36864 % 128 == 0: block stays in one image
    int wave = threadIdx.x >> 5;
    int lane = threadIdx.x & 31;
    int lg   = lane >> 4;        // lane group (K-half selector)
    int ln   = lane & 15;        // row/col within 16

    int p = prem + wave * 16 + ln;   // this lane's pixel (B-matrix column)
    int h = p / W_;
    int w = p % W_;

    v8f acc[4] = {};             // 4 co-tiles x 16x16 f32

    const size_t imgbase = (size_t)n * HP * WP * C_;

    #pragma unroll
    for (int tap = 0; tap < 9; ++tap) {
        int ki = tap / 3, kj = tap % 3;
        // padded coords: interior pixel at (h+1, w+1), tap offset (ki-1, kj-1)
        size_t pix = imgbase + (((size_t)(h + ki) * WP) + (w + kj)) * C_;
        #pragma unroll
        for (int half = 0; half < 2; ++half) {
            int ci0 = half * 32;
            FragBF bhi, blo;
            bhi.q[0] = *(const u32x4*)(s1hi + pix + ci0 + 8 * lg);
            bhi.q[1] = *(const u32x4*)(s1hi + pix + ci0 + 16 + 8 * lg);
            blo.q[0] = *(const u32x4*)(s1lo + pix + ci0 + 8 * lg);
            blo.q[1] = *(const u32x4*)(s1lo + pix + ci0 + 16 + 8 * lg);
            int kk = tap * 64 + ci0;
            #pragma unroll
            for (int cot = 0; cot < 4; ++cot) {
                int co = cot * 16 + ln;  // A-matrix row for this lane
                const unsigned short* whi = wahi + co * K_TOT + kk;
                const unsigned short* wlo = walo + co * K_TOT + kk;
                FragBF ahi, alo;
                ahi.q[0] = *(const u32x4*)(whi + 8 * lg);
                ahi.q[1] = *(const u32x4*)(whi + 16 + 8 * lg);
                alo.q[0] = *(const u32x4*)(wlo + 8 * lg);
                alo.q[1] = *(const u32x4*)(wlo + 16 + 8 * lg);
                acc[cot] = __builtin_amdgcn_wmma_f32_16x16x32_bf16(
                    false, ahi.bf, false, bhi.bf, (short)0, acc[cot], false, false);
                acc[cot] = __builtin_amdgcn_wmma_f32_16x16x32_bf16(
                    false, ahi.bf, false, blo.bf, (short)0, acc[cot], false, false);
                acc[cot] = __builtin_amdgcn_wmma_f32_16x16x32_bf16(
                    false, alo.bf, false, bhi.bf, (short)0, acc[cot], false, false);
            }
        }
    }

    // Epilogue: D layout = lane col ln, VGPR r -> row r + 8*lg. Bias + LeakyReLU.
    #pragma unroll
    for (int cot = 0; cot < 4; ++cot) {
        #pragma unroll
        for (int r = 0; r < 8; ++r) {
            int co = cot * 16 + r + 8 * lg;
            float v = acc[cot][r] + fb[co];
            v = (v >= 0.0f) ? v : 0.2f * v;
            out[((size_t)n * C_ + co) * HW + p] = v;
        }
    }
}

// ---------------------------------------------------------------------------
extern "C" void kernel_launch(void* const* d_in, const int* in_sizes, int n_in,
                              void* d_out, int out_size, void* d_ws, size_t ws_size,
                              hipStream_t stream) {
    const float* x  = (const float*)d_in[0];   // [4,64,192,192]
    const float* y  = (const float*)d_in[1];   // [4,576,192,192]
    const float* fw = (const float*)d_in[2];   // [64,64,3,3]
    const float* fb = (const float*)d_in[3];   // [64]
    float* out = (float*)d_out;

    char* ws = (char*)d_ws;
    const size_t plane_bytes = PLANE_ELEMS * 2;                 // 19,269,632 B
    unsigned short* s1hi = (unsigned short*)(ws);
    unsigned short* s1lo = (unsigned short*)(ws + plane_bytes);
    unsigned short* wahi = (unsigned short*)(ws + 2 * plane_bytes);
    unsigned short* walo = (unsigned short*)(ws + 2 * plane_bytes + (size_t)C_ * K_TOT * 2);

    long long n16 = (long long)((2 * plane_bytes) / 16);
    zero_ws_kernel<<<(unsigned)((n16 + 255) / 256), 256, 0, stream>>>((u32x4*)ws, n16);
    repack_w_kernel<<<(C_ * K_TOT + 255) / 256, 256, 0, stream>>>(fw, wahi, walo);
    dynconv_kernel<<<N_ * H_ * (W_ / 32), 256, 0, stream>>>(x, y, s1hi, s1lo);
    fuseconv_wmma_kernel<<<(N_ * HW) / 128, 256, 0, stream>>>(s1hi, s1lo, wahi, walo, fb, out);
}
